// ContextualTensorProductAttention_64063732187312
// MI455X (gfx1250) — compile-verified, matrix operand order checked by reading, not run on hardware
//
#include <hip/hip_runtime.h>
#include <hip/hip_bf16.h>
#include <math.h>

typedef __bf16 bf16;
typedef __attribute__((ext_vector_type(16))) __bf16 v16bf;
typedef __attribute__((ext_vector_type(8)))  __bf16 v8bf;
typedef __attribute__((ext_vector_type(8)))  float  v8f;
typedef __attribute__((ext_vector_type(4)))  float  v4f;

#define LDS_STRIDE 40   // halves; 80B rows -> 16B aligned, bank-conflict friendly

// ---- gfx1250 async global->LDS staging (guarded; manual fallback) ----------
#if defined(__HIP_DEVICE_COMPILE__) && defined(__gfx1250__) && \
    __has_builtin(__builtin_amdgcn_global_load_async_to_lds_b128)
#define ASYNC_LDS 1
typedef int v4i_g __attribute__((vector_size(16)));            // matches builtin's pointee
typedef __attribute__((address_space(1))) v4i_g* asg_t;        // global src
typedef __attribute__((address_space(3))) v4i_g* asl_t;        // LDS dst
#else
#define ASYNC_LDS 0
#endif

__device__ __forceinline__ void wait_async0() {
#if defined(__HIP_DEVICE_COMPILE__)
#if __has_builtin(__builtin_amdgcn_s_wait_asynccnt)
    __builtin_amdgcn_s_wait_asynccnt(0);
#else
    asm volatile("s_wait_asynccnt 0x0" ::: "memory");
#endif
#endif
}

__device__ __forceinline__ v16bf pack16(v8bf lo, v8bf hi) {
    v16bf r;
#pragma unroll
    for (int i = 0; i < 8; ++i) { r[i] = lo[i]; r[8 + i] = hi[i]; }
    return r;
}

// ---------------------------------------------------------------------------
// Generic batched bf16 WMMA GEMM: C = A * B (or A * B^T if BT).
//   A: row-major MxK (bf16; f32 converted during staging if A_F32)
//   B: row-major KxN (NN) or row-major NxK (BT)
//   C: f32 (or bf16 if C_BF16), row-major MxN
// Block: 256 threads = 8 waves. Block tile 128(M) x 128(N); wave tile 32x64
// (2x4 WMMA 16x16 tiles). K step 32. Double-buffered LDS: next tile is staged
// (async DMA where possible) while WMMAs consume the current tile.
// Batch z decomposed as (z/innerCnt, z%innerCnt) with independent strides.
// ---------------------------------------------------------------------------
template<bool BT, bool A_F32, bool C_BF16>
__global__ __launch_bounds__(256)
void gemm_bf16_wmma(const void* __restrict__ Ap, const bf16* __restrict__ Bp,
                    void* __restrict__ Cp,
                    int M, int N, int K, int lda, int ldb, int ldc,
                    long aO, long aI, long bO, long bI, long cO, long cI,
                    int innerCnt)
{
    const int z  = blockIdx.z;
    const int zo = z / innerCnt, zi = z % innerCnt;
    const long aOff = (long)zo * aO + (long)zi * aI;
    const long bOff = (long)zo * bO + (long)zi * bI;
    const long cOff = (long)zo * cO + (long)zi * cI;

    const int tid  = threadIdx.x;
    const int lane = tid & 31;
    const int wave = tid >> 5;
    const int wM   = wave >> 1;          // 0..3
    const int wN   = wave & 1;           // 0..1
    const int bM   = blockIdx.y * 128;
    const int bN   = blockIdx.x * 128;

    __shared__ bf16 lA[2][128 * LDS_STRIDE];   // 2 x 10240 B
    __shared__ bf16 lB[2][128 * LDS_STRIDE];   // 2 x 10240 B

    v8f acc[2][4];
#pragma unroll
    for (int i = 0; i < 2; ++i)
#pragma unroll
        for (int j = 0; j < 4; ++j)
#pragma unroll
            for (int g = 0; g < 8; ++g) acc[i][j][g] = 0.0f;

    const int mrow = lane & 15;
    const int kb   = (lane >> 4) * 8;    // bf16 A/B fragment K sub-base (ISA 7.12.2)

    constexpr bool anyAsync = (ASYNC_LDS != 0) && (!A_F32 || BT);

    // staging registers (manual paths)
    v4f  aR32[2][2];
    v8bf aR[2];
    v8bf bR[2];

    // ---- load phase: pull next tile (async engine writes LDS[buf] directly)
    auto loadA = [&](int kt, int buf) {
        if constexpr (A_F32) {
            const float* A = (const float*)Ap + aOff;
#pragma unroll
            for (int i = 0; i < 2; ++i) {
                int c = tid + i * 256, r = c >> 2, ch = (c & 3) * 8;
                const float* g = A + (long)(bM + r) * lda + kt + ch;
                aR32[i][0] = *(const v4f*)g;
                aR32[i][1] = *(const v4f*)(g + 4);
            }
        } else {
            const bf16* A = (const bf16*)Ap + aOff;
#if ASYNC_LDS
#pragma unroll
            for (int i = 0; i < 2; ++i) {
                int c = tid + i * 256, r = c >> 2, ch = (c & 3) * 8;
                __builtin_amdgcn_global_load_async_to_lds_b128(
                    (asg_t)(A + (long)(bM + r) * lda + kt + ch),
                    (asl_t)&lA[buf][r * LDS_STRIDE + ch], 0, 0);
            }
#else
#pragma unroll
            for (int i = 0; i < 2; ++i) {
                int c = tid + i * 256, r = c >> 2, ch = (c & 3) * 8;
                aR[i] = *(const v8bf*)(A + (long)(bM + r) * lda + kt + ch);
            }
#endif
            if (kt + 64 < K)   // warm L2 for the tile after next
                __builtin_prefetch(A + (long)(bM + (tid >> 1)) * lda + kt + 64, 0, 1);
        }
        (void)buf;
    };
    auto storeA = [&](int buf) {
        if constexpr (A_F32) {
#pragma unroll
            for (int i = 0; i < 2; ++i) {
                int c = tid + i * 256, r = c >> 2, ch = (c & 3) * 8;
                v8bf h;
#pragma unroll
                for (int j = 0; j < 4; ++j) {
                    h[j]     = (bf16)aR32[i][0][j];
                    h[4 + j] = (bf16)aR32[i][1][j];
                }
                *(v8bf*)&lA[buf][r * LDS_STRIDE + ch] = h;
            }
        } else {
#if !ASYNC_LDS
#pragma unroll
            for (int i = 0; i < 2; ++i) {
                int c = tid + i * 256, r = c >> 2, ch = (c & 3) * 8;
                *(v8bf*)&lA[buf][r * LDS_STRIDE + ch] = aR[i];
            }
#endif
        }
        (void)buf;
    };
    auto loadB = [&](int kt, int buf) {
        const bf16* B = Bp + bOff;
        if constexpr (BT) {   // NxK row-major: rows are K-contiguous -> DMA-able
#if ASYNC_LDS
#pragma unroll
            for (int i = 0; i < 2; ++i) {
                int c = tid + i * 256, r = c >> 2, ch = (c & 3) * 8;
                __builtin_amdgcn_global_load_async_to_lds_b128(
                    (asg_t)(B + (long)(bN + r) * ldb + kt + ch),
                    (asl_t)&lB[buf][r * LDS_STRIDE + ch], 0, 0);
            }
#else
#pragma unroll
            for (int i = 0; i < 2; ++i) {
                int c = tid + i * 256, r = c >> 2, ch = (c & 3) * 8;
                bR[i] = *(const v8bf*)(B + (long)(bN + r) * ldb + kt + ch);
            }
#endif
        } else {              // KxN row-major: read along N, transpose in store
#pragma unroll
            for (int i = 0; i < 2; ++i) {
                int c = tid + i * 256, k = c >> 4, n0 = (c & 15) * 8;
                bR[i] = *(const v8bf*)(B + (long)(kt + k) * ldb + bN + n0);
            }
        }
        (void)buf;
    };
    auto storeB = [&](int buf) {
        if constexpr (BT) {
#if !ASYNC_LDS
#pragma unroll
            for (int i = 0; i < 2; ++i) {
                int c = tid + i * 256, r = c >> 2, ch = (c & 3) * 8;
                *(v8bf*)&lB[buf][r * LDS_STRIDE + ch] = bR[i];
            }
#endif
        } else {
#pragma unroll
            for (int i = 0; i < 2; ++i) {
                int c = tid + i * 256, k = c >> 4, n0 = (c & 15) * 8;
#pragma unroll
                for (int j = 0; j < 8; ++j)
                    lB[buf][(n0 + j) * LDS_STRIDE + k] = bR[i][j];
            }
        }
        (void)buf;
    };

    // ---- prologue: stage K-tile 0 into buffer 0 ----
    loadA(0, 0); loadB(0, 0);
    storeA(0);   storeB(0);
    if constexpr (anyAsync) wait_async0();
    __syncthreads();

    int buf = 0;
    for (int kt = 0; kt < K; kt += 32) {
        const bool more = (kt + 32 < K);
        if (more) { loadA(kt + 32, buf ^ 1); loadB(kt + 32, buf ^ 1); }

        v16bf aFr[2], bFr[4];
#pragma unroll
        for (int t = 0; t < 2; ++t) {
            int r = wM * 32 + t * 16 + mrow;
            aFr[t] = pack16(*(const v8bf*)&lA[buf][r * LDS_STRIDE + kb],
                            *(const v8bf*)&lA[buf][r * LDS_STRIDE + kb + 16]);
        }
#pragma unroll
        for (int t = 0; t < 4; ++t) {
            int r = wN * 64 + t * 16 + mrow;
            bFr[t] = pack16(*(const v8bf*)&lB[buf][r * LDS_STRIDE + kb],
                            *(const v8bf*)&lB[buf][r * LDS_STRIDE + kb + 16]);
        }
#pragma unroll
        for (int i = 0; i < 2; ++i)
#pragma unroll
            for (int j = 0; j < 4; ++j)
                acc[i][j] = __builtin_amdgcn_wmma_f32_16x16x32_bf16(
                    false, aFr[i], false, bFr[j], (short)0, acc[i][j], false, false);

        if (more) {
            storeA(buf ^ 1); storeB(buf ^ 1);
            if constexpr (anyAsync) wait_async0();
        }
        __syncthreads();
        buf ^= 1;
    }

    // ---- epilogue: C/D layout per ISA 7.12.2 (VGPR g -> M=g or 8+g) ----
    const int rBase = bM + wM * 32 + (lane >> 4) * 8;
    const int cBase = bN + wN * 64 + (lane & 15);
#pragma unroll
    for (int i = 0; i < 2; ++i)
#pragma unroll
        for (int j = 0; j < 4; ++j)
#pragma unroll
            for (int g = 0; g < 8; ++g) {
                long idx = (long)(rBase + i * 16 + g) * ldc + (cBase + j * 16);
                if (C_BF16) ((bf16*)Cp + cOff)[idx] = (bf16)acc[i][j][g];
                else        ((float*)Cp + cOff)[idx] = acc[i][j][g];
            }
}

// ---------------------------------------------------------------------------
// f32 -> bf16 converters
// ---------------------------------------------------------------------------
__global__ void cvt_flat(const float* __restrict__ src, bf16* __restrict__ dst, long n)
{
    long i = (long)blockIdx.x * 256 + threadIdx.x;
    if (i < n) dst[i] = (bf16)src[i];
}

__global__ void cvt_submat(const float* __restrict__ src, bf16* __restrict__ dst,
                           int rows, int cols, int dstLd, int dstOff)
{
    long i = (long)blockIdx.x * 256 + threadIdx.x;
    if (i < (long)rows * cols) {
        int r = (int)(i / cols);
        int c = (int)(i % cols);
        dst[(long)r * dstLd + dstOff + c] = (bf16)src[i];
    }
}

// ---------------------------------------------------------------------------
// Per-token: RoPE on B_q/B_k factors + rank contraction -> Q,K,V (bf16).
// P row layout (1408 cols): [Aq 0..95 | Ak 96..111 | Av 112..127 |
//                            Bq 128..895 | Bk 896..1151 | Bv 1152..1407]
// Q scaled by (1/sqrt(128))/6, K and V by 1/2. Output (t, h*128+d).
// ---------------------------------------------------------------------------
__global__ __launch_bounds__(256)
void rope_combine(const float* __restrict__ P,
                  const float* __restrict__ fcos, const float* __restrict__ fsin,
                  bf16* __restrict__ Qb, bf16* __restrict__ Kb, bf16* __restrict__ Vb)
{
    const int t   = blockIdx.x;      // token 0..2047
    const int s   = t & 1023;        // sequence position
    const int tid = threadIdx.x;

    __shared__ float rowP[1408];
    __shared__ float bq[768];
    __shared__ float bk[256];
    __shared__ float lc[64], ls[64];

    const float* pr = P + (long)t * 1408;
    for (int i = tid; i < 1408; i += 256) rowP[i] = pr[i];
    if (tid < 64) { lc[tid] = fcos[s * 64 + tid]; ls[tid] = fsin[s * 64 + tid]; }
    __syncthreads();

#pragma unroll
    for (int i = 0; i < 4; ++i) {
        int e = tid + i * 256;       // 0..1023: 768 Bq elems then 256 Bk elems
        int off, r, d; float* dst;
        if (e < 768) { off = 128; r = e >> 7;        d = e & 127; dst = &bq[e]; }
        else { int e2 = e - 768; off = 896; r = e2 >> 7; d = e2 & 127; dst = &bk[e2]; }
        int j = d & 63;
        float c  = lc[j], sn = ls[j];
        float a  = rowP[off + r * 128 + j];
        float b  = rowP[off + r * 128 + j + 64];
        *dst = (d < 64) ? (a * c - b * sn) : (a * sn + b * c);
    }
    __syncthreads();

    const float qs = 0.014731391f;   // (1/sqrt(128)) / 6
#pragma unroll
    for (int i = 0; i < 8; ++i) {
        int e = tid + i * 256;       // (h,d) pair
        int h = e >> 7, d = e & 127;
        int g = h >> 1;              // kv head
        float q = 0.0f;
#pragma unroll
        for (int r = 0; r < 6; ++r) q += rowP[h * 6 + r] * bq[r * 128 + d];
        q *= qs;
        float k = 0.5f * (rowP[96  + g * 2 + 0] * bk[d] +
                          rowP[96  + g * 2 + 1] * bk[128 + d]);
        float v = 0.5f * (rowP[112 + g * 2 + 0] * rowP[1152 + d] +
                          rowP[112 + g * 2 + 1] * rowP[1280 + d]);
        long o = (long)t * 2048 + e;
        Qb[o] = (bf16)q; Kb[o] = (bf16)k; Vb[o] = (bf16)v;
    }
}

// ---------------------------------------------------------------------------
// In-place masked softmax over 1024-wide rows of probs (layout b,h,q,k).
// ---------------------------------------------------------------------------
__global__ __launch_bounds__(256)
void softmax_rows(float* __restrict__ probs, const float* __restrict__ mask)
{
    const long row = blockIdx.x;               // B*H*S rows
    const int  q   = (int)(row & 1023);
    float* rp = probs + row * 1024;
    const float* mp = mask + (long)q * 1024;
    const int tid = threadIdx.x;
    const int lane = tid & 31, wave = tid >> 5;

    v4f x = *(const v4f*)(rp + tid * 4);
    v4f m = *(const v4f*)(mp + tid * 4);
#pragma unroll
    for (int i = 0; i < 4; ++i) x[i] += m[i];

    float mx = fmaxf(fmaxf(x[0], x[1]), fmaxf(x[2], x[3]));
#pragma unroll
    for (int off = 16; off > 0; off >>= 1) mx = fmaxf(mx, __shfl_xor(mx, off, 32));

    __shared__ float redM[8], redS[8];
    if (lane == 0) redM[wave] = mx;
    __syncthreads();
    mx = redM[0];
#pragma unroll
    for (int i = 1; i < 8; ++i) mx = fmaxf(mx, redM[i]);

    float s = 0.0f;
#pragma unroll
    for (int i = 0; i < 4; ++i) { x[i] = __expf(x[i] - mx); s += x[i]; }
#pragma unroll
    for (int off = 16; off > 0; off >>= 1) s += __shfl_xor(s, off, 32);
    if (lane == 0) redS[wave] = s;
    __syncthreads();
    s = 0.0f;
#pragma unroll
    for (int i = 0; i < 8; ++i) s += redS[i];
    float inv = 1.0f / s;
#pragma unroll
    for (int i = 0; i < 4; ++i) x[i] *= inv;
    *(v4f*)(rp + tid * 4) = x;
}

// ---------------------------------------------------------------------------
extern "C" void kernel_launch(void* const* d_in, const int* in_sizes, int n_in,
                              void* d_out, int out_size, void* d_ws, size_t ws_size,
                              hipStream_t stream)
{
    const float* hidden = (const float*)d_in[0];
    const float* fcos   = (const float*)d_in[1];
    const float* fsin   = (const float*)d_in[2];
    const float* mask   = (const float*)d_in[3];
    const float* WAq    = (const float*)d_in[4];
    const float* WAk    = (const float*)d_in[5];
    const float* WAv    = (const float*)d_in[6];
    const float* WBq    = (const float*)d_in[7];
    const float* WBk    = (const float*)d_in[8];
    const float* WBv    = (const float*)d_in[9];
    const float* Wo     = (const float*)d_in[10];

    const int Bb = 2, S = 1024, D = 2048, H = 16, HD = 128;
    const int T  = Bb * S;        // 2048 tokens
    const int NC = 1408;          // concatenated projection width

    char*  ws  = (char*)d_ws;
    size_t off = 0;
    auto alloc = [&](size_t bytes) -> void* {
        void* p = ws + off; off += (bytes + 255) & ~(size_t)255; return p;
    };
    bf16*  Xb   = (bf16*)alloc((size_t)T * D  * 2);
    bf16*  Wcat = (bf16*)alloc((size_t)D * NC * 2);
    bf16*  Wob  = (bf16*)alloc((size_t)D * D  * 2);
    float* Pm   = (float*)alloc((size_t)T * NC * 4);
    bf16*  Qb   = (bf16*)alloc((size_t)T * 2048 * 2);
    bf16*  Kb   = (bf16*)alloc((size_t)T * 2048 * 2);
    bf16*  Vb   = (bf16*)alloc((size_t)T * 2048 * 2);
    bf16*  Ctxb = (bf16*)alloc((size_t)T * 2048 * 2);

    float* outP  = (float*)d_out;                        // (B,S,D)
    float* probs = outP + (size_t)Bb * S * D;            // (B,H,S,S)

    // 1) f32 -> bf16 conversions (+ weight concatenation)
    cvt_flat<<<(int)(((long)T * D + 255) / 256), 256, 0, stream>>>(hidden, Xb, (long)T * D);
    cvt_submat<<<(D *  96 + 255) / 256, 256, 0, stream>>>(WAq, Wcat, D,  96, NC,    0);
    cvt_submat<<<(D *  16 + 255) / 256, 256, 0, stream>>>(WAk, Wcat, D,  16, NC,   96);
    cvt_submat<<<(D *  16 + 255) / 256, 256, 0, stream>>>(WAv, Wcat, D,  16, NC,  112);
    cvt_submat<<<(D * 768 + 255) / 256, 256, 0, stream>>>(WBq, Wcat, D, 768, NC,  128);
    cvt_submat<<<(D * 256 + 255) / 256, 256, 0, stream>>>(WBk, Wcat, D, 256, NC,  896);
    cvt_submat<<<(D * 256 + 255) / 256, 256, 0, stream>>>(WBv, Wcat, D, 256, NC, 1152);
    cvt_flat<<<(int)(((long)D * D + 255) / 256), 256, 0, stream>>>(Wo, Wob, (long)D * D);

    // 2) Fused projection GEMM: P(T x 1408) = Xb(T x D) @ Wcat(D x 1408)
    {
        dim3 g(NC / 128, T / 128, 1);
        gemm_bf16_wmma<false, false, false><<<g, 256, 0, stream>>>(
            Xb, Wcat, Pm, T, NC, D, D, NC, NC, 0, 0, 0, 0, 0, 0, 1);
    }

    // 3) RoPE + rank contraction -> Q,K,V bf16 (scales folded)
    rope_combine<<<T, 256, 0, stream>>>(Pm, fcos, fsin, Qb, Kb, Vb);

    // 4) scores = Q @ K^T per (b,h); write straight into probs region
    {
        dim3 g(S / 128, S / 128, Bb * H);
        gemm_bf16_wmma<true, false, false><<<g, 256, 0, stream>>>(
            Qb, Kb, probs, S, S, HD, H * HD, H * HD, S,
            (long)S * H * HD, (long)HD,           // A batch strides (b, h)
            (long)S * H * HD, (long)HD,           // B batch strides (b, h)
            (long)H * S * S,  (long)S * S,        // C batch strides (b, h)
            H);
    }

    // 5) masked softmax in place (probs is final output #2)
    softmax_rows<<<Bb * H * S, 256, 0, stream>>>(probs, mask);

    // 6) context = probs @ V per (b,h); A read as f32, converted in staging
    {
        dim3 g(HD / 128, S / 128, Bb * H);
        gemm_bf16_wmma<false, true, true><<<g, 256, 0, stream>>>(
            probs, Vb, Ctxb, S, HD, S, S, H * HD, H * HD,
            (long)H * S * S,  (long)S * S,
            (long)S * H * HD, (long)HD,
            (long)S * H * HD, (long)HD,
            H);
    }

    // 7) output = context @ W_o  (2048 x 2048 x 2048)
    {
        dim3 g(D / 128, T / 128, 1);
        gemm_bf16_wmma<false, false, false><<<g, 256, 0, stream>>>(
            Ctxb, Wob, outP, T, D, D, D, D, D, 0, 0, 0, 0, 0, 0, 1);
    }
}
